// GraphTransformerLayer_47493748359308
// MI455X (gfx1250) — matrix operand order repaired and checked by reading, not compiled, not run on hardware
//
#include <hip/hip_runtime.h>
#include <hip/hip_bf16.h>

// ---------------------------------------------------------------------------
// Graph Transformer layer for gfx1250 (MI455X).
//   * GEMMs: v_wmma_f32_16x16x32_f16, persistent waves, weights pre-packed
//     into B-fragment layout and pinned in VGPRs; A tiles ping-pong
//     double-buffered (manual 2x unroll, no rotation copies) so global-load
//     latency overlaps the WMMA chain.
//   * Edge attention: 1 wave32 per edge, f16 gathers (L2-resident), f32 atomics.
//   * BatchNorm stats folded into GEMM epilogues via per-column atomics.
// ---------------------------------------------------------------------------

typedef __attribute__((ext_vector_type(16))) _Float16 v16h;
typedef __attribute__((ext_vector_type(8)))  float    v8f;

#define IN_DIM  128
#define OUT_DIM 128
#define HID_DIM 256
#define NHEAD   8
#define BN_EPS  1e-5f

// ---------------------- WMMA helpers (ISA 7.12.2 layouts) -------------------

__device__ inline v8f wmma_acc(v16h a, v16h b, v8f c) {
    return __builtin_amdgcn_wmma_f32_16x16x32_f16(
        /*neg_a=*/false, a, /*neg_b=*/false, b,
        /*c_mod=*/(short)0, c, /*reuse_a=*/false, /*reuse_b=*/false);
}

// A fragment: 16x32 f16 tile of A[M x K] (row-major).
// lane<16: elems 0..7 = K kB+hi*8.., elems 8..15 = K kB+16+hi*8..
__device__ inline v16h load_a_frag(const _Float16* __restrict__ A, int lda,
                                   int mBase, int kBase, int lane) {
    int row = lane & 15;
    int hi  = (lane >> 4) & 1;
    const _Float16* p = A + (size_t)(mBase + row) * lda + kBase;
    union { float4 f4; _Float16 h[8]; } u0, u1;
    u0.f4 = *reinterpret_cast<const float4*>(p + hi * 8);       // 16B
    u1.f4 = *reinterpret_cast<const float4*>(p + 16 + hi * 8);  // 16B
    v16h a;
#pragma unroll
    for (int i = 0; i < 8; ++i) { a[i] = u0.h[i]; a[i + 8] = u1.h[i]; }
    return a;
}

template<int NK>
__device__ inline void load_a_tile(v16h (&af)[NK], const _Float16* __restrict__ A,
                                   int mBase, int lane) {
#pragma unroll
    for (int kk = 0; kk < NK; ++kk)
        af[kk] = load_a_frag(A, NK * 32, mBase, kk * 32, lane);
}

// B fragment from pre-packed weights: lane's 16 halves are contiguous (32B).
// Packed layout: [((kk*nct + ct)*32 + lane)*16 + e]
__device__ inline v16h load_b_packed(const _Float16* __restrict__ Wp,
                                     int kk, int ct, int nct, int lane) {
    const _Float16* p = Wp + (((size_t)kk * nct + ct) * 32 + lane) * 16;
    union { float4 f4; _Float16 h[8]; } u0, u1;
    u0.f4 = *reinterpret_cast<const float4*>(p);
    u1.f4 = *reinterpret_cast<const float4*>(p + 8);
    v16h b;
#pragma unroll
    for (int i = 0; i < 8; ++i) { b[i] = u0.h[i]; b[i + 8] = u1.h[i]; }
    return b;
}

// Pack f32 weights W[K x NC] (row-major) into the B-fragment layout above.
// Fragment semantics: col=ct*16+(lane&15), hi=lane>>4,
//                     K = kk*32 + hi*8 + (e&7) + (e>>3)*16   (e=0..15)
__global__ void k_pack_b(const float* __restrict__ W, _Float16* __restrict__ out,
                         int K, int NC) {
    int total = K * NC;
    int nct = NC >> 4;
    for (int idx = blockIdx.x * blockDim.x + threadIdx.x; idx < total;
         idx += gridDim.x * blockDim.x) {
        int e    = idx & 15;
        int lane = (idx >> 4) & 31;
        int rest = idx >> 9;              // kk*nct + ct
        int ct   = rest % nct;
        int kk   = rest / nct;
        int col  = ct * 16 + (lane & 15);
        int krow = kk * 32 + ((lane >> 4) ? 8 : 0) + (e & 7) + ((e >> 3) << 4);
        out[idx] = (_Float16)W[(size_t)krow * NC + col];
    }
}

// ---------------------------- elementwise kernels ---------------------------

union H4 { uint2 u; _Float16 h[4]; };

__global__ void k_f32_to_f16_v4(const float4* __restrict__ in, uint2* __restrict__ out,
                                long long n4) {
    long long i = (long long)blockIdx.x * blockDim.x + threadIdx.x;
    long long stride = (long long)gridDim.x * blockDim.x;
    for (; i < n4; i += stride) {
        float4 v = in[i];
        H4 o;
        o.h[0] = (_Float16)v.x; o.h[1] = (_Float16)v.y;
        o.h[2] = (_Float16)v.z; o.h[3] = (_Float16)v.w;
        out[i] = o.u;
    }
}

// attn = wV / z (per-head divisor), stored f16 for the Wo GEMM.
__global__ void k_attn_div(const float4* __restrict__ wV, const float* __restrict__ z,
                           uint2* __restrict__ attn_h, long long total4) {
    long long i = (long long)blockIdx.x * blockDim.x + threadIdx.x;
    long long stride = (long long)gridDim.x * blockDim.x;
    for (; i < total4; i += stride) {
        int cb = (int)((i << 2) & 127);            // 4 cols share one head
        long long row = i >> 5;                    // (i*4) >> 7
        float invz = 1.0f / z[row * NHEAD + (cb >> 4)];
        float4 v = wV[i];
        H4 o;
        o.h[0] = (_Float16)(v.x * invz); o.h[1] = (_Float16)(v.y * invz);
        o.h[2] = (_Float16)(v.z * invz); o.h[3] = (_Float16)(v.w * invz);
        attn_h[i] = o.u;
    }
}

// BatchNorm apply from accumulated per-column sum/sumsq.
__global__ void k_bn_apply(const float4* __restrict__ x, const float* __restrict__ stats,
                           const float* __restrict__ gamma, const float* __restrict__ beta,
                           float4* __restrict__ out_f32, uint2* __restrict__ out_f16,
                           float invN, long long total4) {
    long long i = (long long)blockIdx.x * blockDim.x + threadIdx.x;
    long long stride = (long long)gridDim.x * blockDim.x;
    for (; i < total4; i += stride) {
        int cb = (int)((i << 2) & 127);
        float4 v = x[i];
        float r[4] = {v.x, v.y, v.z, v.w};
#pragma unroll
        for (int j = 0; j < 4; ++j) {
            int c = cb + j;
            float mu  = stats[c] * invN;
            float var = fmaxf(stats[128 + c] * invN - mu * mu, 0.0f);
            r[j] = (r[j] - mu) * rsqrtf(var + BN_EPS) * gamma[c] + beta[c];
        }
        out_f32[i] = make_float4(r[0], r[1], r[2], r[3]);
        if (out_f16) {
            H4 o;
            o.h[0] = (_Float16)r[0]; o.h[1] = (_Float16)r[1];
            o.h[2] = (_Float16)r[2]; o.h[3] = (_Float16)r[3];
            out_f16[i] = o.u;
        }
    }
}

// ------------------------------- GEMM kernels -------------------------------
// Persistent waves: each wave owns one 16-col tile, keeps its NK B-fragments
// in VGPRs, ping-pong double-buffers A tiles across the M grid-stride loop.

// Compute one tile and store f16 (optional bias+relu).
template<int NK, int NC, bool BIASRELU>
__device__ inline void tile_store_f16(const v16h (&af)[NK], const v16h (&bf)[NK],
                                      _Float16* __restrict__ O, int mBase,
                                      int ct, int col, int rOff, float b) {
    v8f acc = {};
#pragma unroll
    for (int kk = 0; kk < NK; ++kk) acc = wmma_acc(af[kk], bf[kk], acc);
    _Float16* o = O + (size_t)(mBase + rOff) * NC + ct * 16 + col;
#pragma unroll
    for (int r = 0; r < 8; ++r) {
        float v = acc[r];
        if (BIASRELU) v = fmaxf(v + b, 0.0f);
        o[(size_t)r * NC] = (_Float16)v;
    }
}

// O = A(N x NK*32) * W -> f16, optional bias (+relu).  grid.y*8 col groups.
template<int NK, int NC, bool BIASRELU>
__global__ void k_gemm_f16out(const _Float16* __restrict__ A,
                              const _Float16* __restrict__ Wp,
                              const float* __restrict__ bias,
                              _Float16* __restrict__ O, int mtiles) {
    constexpr int NCT = NC / 16;
    int lane = threadIdx.x & 31;
    int wave = threadIdx.x >> 5;
    int ct   = blockIdx.y * 8 + wave;
    v16h bf[NK];
#pragma unroll
    for (int kk = 0; kk < NK; ++kk) bf[kk] = load_b_packed(Wp, kk, ct, NCT, lane);
    int col  = lane & 15;
    int rOff = (lane >> 4) * 8;
    float b = BIASRELU ? bias[ct * 16 + col] : 0.0f;

    const int stride = gridDim.x;
    int mt = blockIdx.x;
    v16h af0[NK], af1[NK];
    if (mt < mtiles) load_a_tile<NK>(af0, A, mt * 16, lane);
    while (mt < mtiles) {
        int mt1 = mt + stride;
        if (mt1 < mtiles) load_a_tile<NK>(af1, A, mt1 * 16, lane);   // prefetch
        tile_store_f16<NK, NC, BIASRELU>(af0, bf, O, mt * 16, ct, col, rOff, b);
        if (mt1 >= mtiles) break;
        int mt2 = mt1 + stride;
        if (mt2 < mtiles) load_a_tile<NK>(af0, A, mt2 * 16, lane);   // prefetch
        tile_store_f16<NK, NC, BIASRELU>(af1, bf, O, mt1 * 16, ct, col, rOff, b);
        mt = mt2;
    }
}

// Compute one tile: out = acc + bias + resid (f32) and BN-stat atomics.
template<int NK>
__device__ inline void tile_store_f32res(const v16h (&af)[NK], const v16h (&bf)[NK],
                                         const float* __restrict__ resid,
                                         float* __restrict__ out,
                                         float* __restrict__ stats, int mBase,
                                         int ct, int col, int rOff, int lane, float b) {
    v8f acc = {};
#pragma unroll
    for (int kk = 0; kk < NK; ++kk) acc = wmma_acc(af[kk], bf[kk], acc);
    size_t base = (size_t)(mBase + rOff) * OUT_DIM + ct * 16 + col;
    float s = 0.f, s2 = 0.f;
#pragma unroll
    for (int r = 0; r < 8; ++r) {
        float v = acc[r] + b + resid[base + (size_t)r * OUT_DIM];
        out[base + (size_t)r * OUT_DIM] = v;
        s += v; s2 += v * v;
    }
    s  += __shfl_down(s, 16, 32);
    s2 += __shfl_down(s2, 16, 32);
    if (lane < 16) {
        atomicAdd(&stats[ct * 16 + lane], s);
        atomicAdd(&stats[128 + ct * 16 + lane], s2);
    }
}

// out = resid + A(N x NK*32) * W + bias -> f32, accumulating BN stats.
template<int NK>
__global__ void k_gemm_f32res(const _Float16* __restrict__ A,
                              const _Float16* __restrict__ Wp,
                              const float* __restrict__ bias,
                              const float* __restrict__ resid,
                              float* __restrict__ out, float* __restrict__ stats,
                              int mtiles) {
    int lane = threadIdx.x & 31;
    int wave = threadIdx.x >> 5;
    int ct   = wave;                                   // NC == 128 -> 8 tiles
    v16h bf[NK];
#pragma unroll
    for (int kk = 0; kk < NK; ++kk) bf[kk] = load_b_packed(Wp, kk, ct, 8, lane);
    int col  = lane & 15;
    int rOff = (lane >> 4) * 8;
    float b = bias[ct * 16 + col];

    const int stride = gridDim.x;
    int mt = blockIdx.x;
    v16h af0[NK], af1[NK];
    if (mt < mtiles) load_a_tile<NK>(af0, A, mt * 16, lane);
    while (mt < mtiles) {
        int mt1 = mt + stride;
        if (mt1 < mtiles) load_a_tile<NK>(af1, A, mt1 * 16, lane);   // prefetch
        tile_store_f32res<NK>(af0, bf, resid, out, stats, mt * 16, ct, col, rOff, lane, b);
        if (mt1 >= mtiles) break;
        int mt2 = mt1 + stride;
        if (mt2 < mtiles) load_a_tile<NK>(af0, A, mt2 * 16, lane);   // prefetch
        tile_store_f32res<NK>(af1, bf, resid, out, stats, mt1 * 16, ct, col, rOff, lane, b);
        mt = mt2;
    }
}

// ------------------------------- edge kernel --------------------------------
// One wave32 per edge; lane holds 4 contiguous f16 channels (8B coalesced).
// Head = lane>>2; butterfly shfl_xor reduces the 16-dim dot over 4 lanes.
__global__ void k_edge(const _Float16* __restrict__ Qh, const _Float16* __restrict__ Kh,
                       const _Float16* __restrict__ Vh, const int* __restrict__ src,
                       const int* __restrict__ dst, float* __restrict__ wV,
                       float* __restrict__ z, int E) {
    int wave = blockIdx.x * (blockDim.x >> 5) + (threadIdx.x >> 5);
    int lane = threadIdx.x & 31;
    if (wave >= E) return;
    int s = src[wave];
    int d = dst[wave];
    int i0 = lane * 4;
    H4 kk, qq, vv;
    kk.u = *reinterpret_cast<const uint2*>(Kh + (size_t)s * OUT_DIM + i0);
    qq.u = *reinterpret_cast<const uint2*>(Qh + (size_t)d * OUT_DIM + i0);
    vv.u = *reinterpret_cast<const uint2*>(Vh + (size_t)s * OUT_DIM + i0);
    float p = 0.f;
#pragma unroll
    for (int j = 0; j < 4; ++j) p += (float)kk.h[j] * (float)qq.h[j];
    p += __shfl_xor(p, 1, 32);
    p += __shfl_xor(p, 2, 32);
    float sc = __expf(fminf(fmaxf(p * 0.25f, -5.0f), 5.0f));   // /sqrt(16)
    float* wvp = wV + (size_t)d * OUT_DIM + i0;
#pragma unroll
    for (int j = 0; j < 4; ++j) atomicAdd(wvp + j, sc * (float)vv.h[j]);
    if ((lane & 3) == 0) atomicAdd(z + (size_t)d * NHEAD + (lane >> 2), sc);
}

// -------------------------------- launcher ----------------------------------

extern "C" void kernel_launch(void* const* d_in, const int* in_sizes, int n_in,
                              void* d_out, int out_size, void* d_ws, size_t ws_size,
                              hipStream_t stream) {
    const float* h      = (const float*)d_in[0];
    const float* Wq     = (const float*)d_in[1];
    const float* Wk     = (const float*)d_in[2];
    const float* Wv     = (const float*)d_in[3];
    const float* Wo     = (const float*)d_in[4];
    const float* bo     = (const float*)d_in[5];
    const float* gamma1 = (const float*)d_in[6];
    const float* beta1  = (const float*)d_in[7];
    const float* Wf1    = (const float*)d_in[8];
    const float* bf1    = (const float*)d_in[9];
    const float* Wf2    = (const float*)d_in[10];
    const float* bf2    = (const float*)d_in[11];
    const float* gamma2 = (const float*)d_in[12];
    const float* beta2  = (const float*)d_in[13];
    const int*   src    = (const int*)d_in[14];
    const int*   dst    = (const int*)d_in[15];

    const int N = in_sizes[0] / IN_DIM;        // 100000 (multiple of 16)
    const int E = in_sizes[14];                // 1600000 (multiple of 8)
    const long long NF = (long long)N * OUT_DIM;
    const int mtiles = N / 16;                 // 6250

    // -------- workspace carve-out (256B aligned) --------
    size_t off = 0;
    auto alloc = [&](size_t bytes) -> char* {
        char* p = (char*)d_ws + off;
        off = (off + bytes + 255) & ~(size_t)255;
        return p;
    };
    _Float16* hf     = (_Float16*)alloc((size_t)NF * 2);
    _Float16* Qh     = (_Float16*)alloc((size_t)NF * 2);
    _Float16* Kh     = (_Float16*)alloc((size_t)NF * 2);
    _Float16* Vh     = (_Float16*)alloc((size_t)NF * 2);          // reused: attn_h
    _Float16* t16    = (_Float16*)alloc((size_t)N * HID_DIM * 2); // FFN hidden
    float*    wV     = (float*)alloc((size_t)NF * 4);             // reused: xbn f32
    float*    xbuf   = (float*)alloc((size_t)NF * 4);             // pre-BN x / x2
    float*    zacc   = (float*)alloc((size_t)N * NHEAD * 4);
    _Float16* Wq16   = (_Float16*)alloc(IN_DIM * OUT_DIM * 2);    // packed
    _Float16* Wk16   = (_Float16*)alloc(IN_DIM * OUT_DIM * 2);
    _Float16* Wv16   = (_Float16*)alloc(IN_DIM * OUT_DIM * 2);
    _Float16* Wo16   = (_Float16*)alloc(OUT_DIM * OUT_DIM * 2);
    _Float16* Wf1_16 = (_Float16*)alloc(OUT_DIM * HID_DIM * 2);
    _Float16* Wf2_16 = (_Float16*)alloc(HID_DIM * OUT_DIM * 2);
    float*    stats1 = (float*)alloc(256 * 4);                    // sum | sumsq
    float*    stats2 = (float*)alloc(256 * 4);
    _Float16* xh     = Qh;     // BN1 f16 output reuses Qh after edge phase
    _Float16* attn_h = Vh;     // attn f16 reuses Vh after edge phase

    hipMemsetAsync(wV, 0, (size_t)NF * 4, stream);
    hipMemsetAsync(zacc, 0, (size_t)N * NHEAD * 4, stream);
    hipMemsetAsync(stats1, 0, 256 * 4, stream);
    hipMemsetAsync(stats2, 0, 256 * 4, stream);

    // -------- convert h, pack weights into B-fragment layout --------
    k_f32_to_f16_v4<<<2048, 256, 0, stream>>>((const float4*)h, (uint2*)hf, NF >> 2);
    k_pack_b<<<64, 256, 0, stream>>>(Wq, Wq16, IN_DIM, OUT_DIM);
    k_pack_b<<<64, 256, 0, stream>>>(Wk, Wk16, IN_DIM, OUT_DIM);
    k_pack_b<<<64, 256, 0, stream>>>(Wv, Wv16, IN_DIM, OUT_DIM);
    k_pack_b<<<64, 256, 0, stream>>>(Wo, Wo16, OUT_DIM, OUT_DIM);
    k_pack_b<<<128, 256, 0, stream>>>(Wf1, Wf1_16, OUT_DIM, HID_DIM);
    k_pack_b<<<128, 256, 0, stream>>>(Wf2, Wf2_16, HID_DIM, OUT_DIM);

    const int gx = 1024;   // persistent blocks per GEMM

    // -------- Q, K, V projections (WMMA, persistent) --------
    k_gemm_f16out<4, 128, false><<<dim3(gx, 1), 256, 0, stream>>>(hf, Wq16, nullptr, Qh, mtiles);
    k_gemm_f16out<4, 128, false><<<dim3(gx, 1), 256, 0, stream>>>(hf, Wk16, nullptr, Kh, mtiles);
    k_gemm_f16out<4, 128, false><<<dim3(gx, 1), 256, 0, stream>>>(hf, Wv16, nullptr, Vh, mtiles);

    // -------- edge attention scatter (1 wave per edge) --------
    k_edge<<<(E + 7) / 8, 256, 0, stream>>>(Qh, Kh, Vh, src, dst, wV, zacc, E);

    // -------- attn = wV / z --------
    k_attn_div<<<2048, 256, 0, stream>>>((const float4*)wV, zacc, (uint2*)attn_h, NF >> 2);

    // -------- x = h + attn @ Wo + bo, with BN1 stats --------
    k_gemm_f32res<4><<<dim3(gx), 256, 0, stream>>>(attn_h, Wo16, bo, h, xbuf, stats1, mtiles);

    // -------- BN1 apply: f32 (residual) + f16 (FFN input) --------
    k_bn_apply<<<2048, 256, 0, stream>>>((const float4*)xbuf, stats1, gamma1, beta1,
                                         (float4*)wV, (uint2*)xh, 1.0f / (float)N, NF >> 2);

    // -------- FFN1: relu(xbn @ Wf1 + bf1) --------
    k_gemm_f16out<4, 256, true><<<dim3(gx, 2), 256, 0, stream>>>(xh, Wf1_16, bf1, t16, mtiles);

    // -------- FFN2: x2 = xbn + t @ Wf2 + bf2, with BN2 stats --------
    k_gemm_f32res<8><<<dim3(gx), 256, 0, stream>>>(t16, Wf2_16, bf2, wV, xbuf, stats2, mtiles);

    // -------- BN2 apply -> d_out (f32) --------
    k_bn_apply<<<2048, 256, 0, stream>>>((const float4*)xbuf, stats2, gamma2, beta2,
                                         (float4*)d_out, (uint2*)nullptr,
                                         1.0f / (float)N, NF >> 2);
}
